// GCN_57535381897259
// MI455X (gfx1250) — matrix operand order repaired and checked by reading, not compiled, not run on hardware
//
#include <hip/hip_runtime.h>
#include <hip/hip_bf16.h>

// ---------------------------------------------------------------------------
// Types for CDNA5 WMMA
// ---------------------------------------------------------------------------
typedef __attribute__((ext_vector_type(16))) __bf16 v16bf;
typedef __attribute__((ext_vector_type(8)))  __bf16 v8bf;
typedef __attribute__((ext_vector_type(8)))  float  v8f;

// ---------------------------------------------------------------------------
// Degree / normalization kernels
// ---------------------------------------------------------------------------
__global__ void deg_init_kernel(float* deg, int n) {
    int i = blockIdx.x * blockDim.x + threadIdx.x;
    if (i < n) deg[i] = 1.0f;  // self loop contributes 1
}

__global__ void deg_count_kernel(const int* __restrict__ dst, float* deg, int E) {
    int i = blockIdx.x * blockDim.x + threadIdx.x;
    if (i < E) atomicAdd(&deg[dst[i]], 1.0f);
}

__global__ void deg_finalize_kernel(float* deg, int n) {
    int i = blockIdx.x * blockDim.x + threadIdx.x;
    if (i < n) deg[i] = rsqrtf(deg[i]);  // deg >= 1 always
}

// ---------------------------------------------------------------------------
// fp32 -> bf16 row-major copy (A operand prep)
// ---------------------------------------------------------------------------
__global__ void cvt_bf16_kernel(const float* __restrict__ in,
                                __bf16* __restrict__ out, long long n) {
    long long i = (long long)blockIdx.x * blockDim.x + threadIdx.x;
    if (i < n) out[i] = (__bf16)in[i];
}

// ---------------------------------------------------------------------------
// W[K,N] fp32 -> Wt[Npad,K] bf16 transposed + zero-padded (B operand prep).
// Wt row c holds column c of W contiguously in K -> exactly the per-lane
// 16-consecutive-K layout this GEMM's B loads use (same k order as A).
// ---------------------------------------------------------------------------
__global__ void transpose_pad_bf16_kernel(const float* __restrict__ W,
                                          __bf16* __restrict__ Wt,
                                          int K, int N, int Npad) {
    int idx = blockIdx.x * blockDim.x + threadIdx.x;
    if (idx >= Npad * K) return;
    int c = idx / K;
    int k = idx - c * K;
    float v = (c < N) ? W[(size_t)k * N + c] : 0.0f;
    Wt[(size_t)c * K + k] = (__bf16)v;
}

// ---------------------------------------------------------------------------
// WMMA GEMM: C[M,Nout] = A[M,K](bf16) @ Wt[Npad,K](bf16,transposed), fp32 out.
// Wave computes a 16x(16*NT) strip: one A fragment reused over NT B tiles,
// NT independent accumulators per wave. 8 waves/block stack along M.
// K compile-time, multiple of 32. M multiple of 16. Store guarded by Nout.
// ---------------------------------------------------------------------------
template <int K, int NT>
__global__ void wmma_gemm_kernel(const __bf16* __restrict__ A,
                                 const __bf16* __restrict__ Wt,
                                 float* __restrict__ C,
                                 int Nout, int tilesM) {
    const int lane = threadIdx.x & 31;
    const int wave = threadIdx.x >> 5;
    const int tm   = blockIdx.x * 8 + wave;  // M-tile index (wave-uniform)
    if (tm >= tilesM) return;                // uniform per wave -> EXEC all 1s

    const int l15  = lane & 15;
    const int half = lane >> 4;              // 0: K-low half, 1: K-high half
    const int row  = tm * 16 + l15;          // A row for this lane
    const int cbase = blockIdx.y * (16 * NT);

    // Per-lane base pointers (K-offsets advance by 32 each step)
    const __bf16* aRow = A + (size_t)row * K;
    const __bf16* bCol[NT];
#pragma unroll
    for (int j = 0; j < NT; ++j)
        bCol[j] = Wt + (size_t)(cbase + j * 16 + l15) * K + half * 16;

    v8f acc[NT];
#pragma unroll
    for (int j = 0; j < NT; ++j) acc[j] = (v8f){};

#pragma unroll 4
    for (int k0 = 0; k0 < K; k0 += 32) {
        // A fragment: elems 0..15 <- A[row][k0+half*16 ..] (16 K-consecutive)
        v8bf alo = *(const v8bf*)(aRow + k0 + half * 16);
        v8bf ahi = *(const v8bf*)(aRow + k0 + half * 16 + 8);
        v16bf a = __builtin_shufflevector(alo, ahi,
                  0, 1, 2, 3, 4, 5, 6, 7, 8, 9, 10, 11, 12, 13, 14, 15);
#pragma unroll
        for (int j = 0; j < NT; ++j) {
            // B fragment: 16 K-consecutive bf16 of column (cbase+j*16+l15)
            v16bf b = *(const v16bf*)(bCol[j] + k0);
            acc[j] = __builtin_amdgcn_wmma_f32_16x16x32_bf16(
                false, a, false, b, (short)0, acc[j], false, false);
        }
    }

    // D layout: element v -> row tm*16 + half*8 + v, col cbase + j*16 + l15
#pragma unroll
    for (int j = 0; j < NT; ++j) {
        int col = cbase + j * 16 + l15;
        if (col < Nout) {
#pragma unroll
            for (int v = 0; v < 8; ++v) {
                int m = tm * 16 + half * 8 + v;
                C[(size_t)m * Nout + col] = acc[j][v];
            }
        }
    }
}

// ---------------------------------------------------------------------------
// out[i,f] = bias[f] + dis[i]^2 * H[i,f]   (self-loop term + bias)
// ---------------------------------------------------------------------------
__global__ void agg_init_kernel(const float* __restrict__ H,
                                const float* __restrict__ dis,
                                const float* __restrict__ bias,
                                float* __restrict__ out, int n, int F) {
    int idx = blockIdx.x * blockDim.x + threadIdx.x;
    if (idx >= n * F) return;
    int i = idx / F;
    int f = idx - i * F;
    float d = dis[i];
    out[idx] = bias[f] + d * d * H[idx];
}

// ---------------------------------------------------------------------------
// Edge scatter: out[dst] += dis[src]*dis[dst] * H[src]; TPE threads/edge,
// one float4 per thread.
// ---------------------------------------------------------------------------
__global__ void edge_agg_kernel(const float* __restrict__ H,
                                const int* __restrict__ srcs,
                                const int* __restrict__ dsts,
                                const float* __restrict__ dis,
                                float* __restrict__ out,
                                int E, int F, int TPE) {
    long long idx = (long long)blockIdx.x * blockDim.x + threadIdx.x;
    long long tot = (long long)E * TPE;
    if (idx >= tot) return;
    int e    = (int)(idx / TPE);
    int slot = (int)(idx - (long long)e * TPE);
    int s = srcs[e];
    int d = dsts[e];
    float w = dis[s] * dis[d];
    const float4 hv = *reinterpret_cast<const float4*>(H + (size_t)s * F + slot * 4);
    float* op = out + (size_t)d * F + slot * 4;
    atomicAdd(op + 0, hv.x * w);
    atomicAdd(op + 1, hv.y * w);
    atomicAdd(op + 2, hv.z * w);
    atomicAdd(op + 3, hv.w * w);
}

// ---------------------------------------------------------------------------
// ReLU + bf16 convert: H1bf = bf16(max(H1, 0))   (GEMM2 A-operand prep)
// ---------------------------------------------------------------------------
__global__ void relu_cvt_kernel(const float* __restrict__ in,
                                __bf16* __restrict__ out, long long n) {
    long long i = (long long)blockIdx.x * blockDim.x + threadIdx.x;
    if (i < n) out[i] = (__bf16)fmaxf(in[i], 0.0f);
}

// ---------------------------------------------------------------------------
// Row-wise log_softmax over 40 classes; one wave (32 lanes) per row.
// ---------------------------------------------------------------------------
__global__ void log_softmax40_kernel(float* __restrict__ out, int n) {
    int row  = blockIdx.x * (blockDim.x >> 5) + (threadIdx.x >> 5);
    int lane = threadIdx.x & 31;
    if (row >= n) return;
    float* p = out + (size_t)row * 40;
    float a = p[lane];                               // lanes 0..31
    float b = (lane < 8) ? p[lane + 32] : -INFINITY; // lanes 0..7 -> 32..39
    float m = fmaxf(a, b);
#pragma unroll
    for (int off = 16; off; off >>= 1) m = fmaxf(m, __shfl_xor(m, off));
    float ea = expf(a - m);
    float eb = (lane < 8) ? expf(b - m) : 0.0f;
    float s = ea + eb;
#pragma unroll
    for (int off = 16; off; off >>= 1) s += __shfl_xor(s, off);
    float ls = logf(s);
    p[lane] = a - m - ls;
    if (lane < 8) p[lane + 32] = b - m - ls;
}

// ---------------------------------------------------------------------------
// Orchestration
// ---------------------------------------------------------------------------
extern "C" void kernel_launch(void* const* d_in, const int* in_sizes, int n_in,
                              void* d_out, int out_size, void* d_ws, size_t ws_size,
                              hipStream_t stream) {
    const float* x  = (const float*)d_in[0];   // [N, 512]
    const float* W1 = (const float*)d_in[1];   // [512, 256]
    const float* b1 = (const float*)d_in[2];   // [256]
    const float* W2 = (const float*)d_in[3];   // [256, 40]
    const float* b2 = (const float*)d_in[4];   // [40]
    const int*   ei = (const int*)d_in[5];     // [2, E]

    const int NFEAT = 512, NHID = 256, NCLASS = 40, NCPAD = 48;
    const int n = in_sizes[0] / NFEAT;         // 100000
    const int E = in_sizes[5] / 2;             // 3200000
    const int* src = ei;
    const int* dst = ei + E;

    // ---- workspace carving (256B aligned) ----
    char* ws = (char*)d_ws;
    size_t off = 0;
    auto carve = [&](size_t bytes) {
        char* p = ws + off;
        off += (bytes + 255) & ~(size_t)255;
        return p;
    };
    float*  dis  = (float*)carve((size_t)n * sizeof(float));
    __bf16* xbf  = (__bf16*)carve((size_t)n * NFEAT * sizeof(__bf16));
    __bf16* W1t  = (__bf16*)carve((size_t)NHID * NFEAT * sizeof(__bf16));
    __bf16* W2t  = (__bf16*)carve((size_t)NCPAD * NHID * sizeof(__bf16));
    float*  XW1  = (float*)carve((size_t)n * NHID * sizeof(float));
    float*  H1   = (float*)carve((size_t)n * NHID * sizeof(float));
    __bf16* H1bf = (__bf16*)carve((size_t)n * NHID * sizeof(__bf16));
    float*  HW2  = (float*)carve((size_t)n * NCLASS * sizeof(float));
    float*  OUT  = (float*)d_out;              // [n, 40]

    const int B = 256;
    const int tilesM = n / 16;                 // 6250

    // 1) degree / normalization
    deg_init_kernel<<<(n + B - 1) / B, B, 0, stream>>>(dis, n);
    deg_count_kernel<<<(E + B - 1) / B, B, 0, stream>>>(dst, dis, E);
    deg_finalize_kernel<<<(n + B - 1) / B, B, 0, stream>>>(dis, n);

    // 2) operand prep: x -> bf16, W1/W2 -> transposed padded bf16
    {
        long long tot = (long long)n * NFEAT;
        cvt_bf16_kernel<<<(unsigned)((tot + B - 1) / B), B, 0, stream>>>(x, xbf, tot);
        int t1 = NHID * NFEAT;
        transpose_pad_bf16_kernel<<<(t1 + B - 1) / B, B, 0, stream>>>(W1, W1t, NFEAT, NHID, NHID);
        int t2 = NCPAD * NHID;
        transpose_pad_bf16_kernel<<<(t2 + B - 1) / B, B, 0, stream>>>(W2, W2t, NHID, NCLASS, NCPAD);
    }

    // 3) GEMM1: XW1 = x @ W1   (M=100000, K=512, N=256; wave strip 16x64)
    {
        dim3 grid((tilesM + 7) / 8, NHID / 64);   // (782, 4)
        wmma_gemm_kernel<512, 4><<<grid, B, 0, stream>>>(xbf, W1t, XW1, NHID, tilesM);
    }

    // 4) layer-1 aggregation: H1 = b1 + dis^2*XW1 + edge scatter; ReLU->bf16
    {
        long long tot = (long long)n * NHID;
        agg_init_kernel<<<(unsigned)((tot + B - 1) / B), B, 0, stream>>>(XW1, dis, b1, H1, n, NHID);
        int TPE = NHID / 4;                       // 64 threads per edge
        long long etot = (long long)E * TPE;
        edge_agg_kernel<<<(unsigned)((etot + B - 1) / B), B, 0, stream>>>(XW1, src, dst, dis, H1, E, NHID, TPE);
        relu_cvt_kernel<<<(unsigned)((tot + B - 1) / B), B, 0, stream>>>(H1, H1bf, tot);
    }

    // 5) GEMM2: HW2 = H1 @ W2   (M=100000, K=256, N=40 padded to 48; strip 16x48)
    {
        dim3 grid((tilesM + 7) / 8, 1);
        wmma_gemm_kernel<256, 3><<<grid, B, 0, stream>>>(H1bf, W2t, HW2, NCLASS, tilesM);
    }

    // 6) layer-2 aggregation directly into d_out
    {
        long long tot = (long long)n * NCLASS;
        agg_init_kernel<<<(unsigned)((tot + B - 1) / B), B, 0, stream>>>(HW2, dis, b2, OUT, n, NCLASS);
        int TPE = NCLASS / 4;                     // 10 threads per edge
        long long etot = (long long)E * TPE;
        edge_agg_kernel<<<(unsigned)((etot + B - 1) / B), B, 0, stream>>>(HW2, src, dst, dis, OUT, E, NCLASS, TPE);
    }

    // 7) log_softmax in place, 8 rows per 256-thread block
    {
        int rowsPerBlock = B / 32;
        log_softmax40_kernel<<<(n + rowsPerBlock - 1) / rowsPerBlock, B, 0, stream>>>(OUT, n);
    }
}
